// MultiHeadAttention_2748779070010
// MI455X (gfx1250) — matrix operand order repaired and checked by reading, not compiled, not run on hardware
//
#include <hip/hip_runtime.h>
#include <hip/hip_bf16.h>
#include <stdint.h>

// ---------------------------------------------------------------------------
// Types for CDNA5 WMMA (wave32): v_wmma_f32_16x16x32_bf16
//   A: 16x32 bf16  -> 16 bf16/lane (8 VGPRs)
//   B: 32x16 bf16  -> 16 bf16/lane (8 VGPRs)
//   C/D: 16x16 f32 -> 8 f32/lane
// ---------------------------------------------------------------------------
typedef __attribute__((ext_vector_type(16))) __bf16 bf16x16;
typedef __attribute__((ext_vector_type(8)))  float  floatx8;

union FragBF {
    bf16x16        bf;
    uint4          q[2];
    unsigned short h16[16];
};

__device__ __forceinline__ unsigned short f2bf(float f) {
    unsigned int u = __float_as_uint(f);
    u += 0x7fffu + ((u >> 16) & 1u);   // round-to-nearest-even
    return (unsigned short)(u >> 16);
}

// LDS offset of a __shared__ object: low 32 bits of its flat address
// (flat LDS aperture: addr[31:0] == LDS byte offset).
__device__ __forceinline__ unsigned lds_off(const void* p) {
    return (unsigned)(uintptr_t)p;
}

// Async memory -> LDS copy of 16 bytes per lane (CDNA5, ASYNCcnt tracked).
__device__ __forceinline__ void async_copy_b128(unsigned lds_addr, const void* gptr) {
    asm volatile("global_load_async_to_lds_b128 %0, %1, off"
                 :: "v"(lds_addr), "v"(gptr) : "memory");
}
__device__ __forceinline__ void wait_async0() {
    asm volatile("s_wait_asynccnt 0x0" ::: "memory");
}

#define S_LEN  2048
#define BATCH  2
#define DMODEL 1024
#define NHEAD  16
#define DK     64

// ---------------------------------------------------------------------------
// f32 -> bf16 elementwise convert
// ---------------------------------------------------------------------------
__global__ void cvt_bf16_kernel(const float* __restrict__ src,
                                unsigned short* __restrict__ dst, int n) {
    int i = blockIdx.x * blockDim.x + threadIdx.x;
    if (i < n) dst[i] = f2bf(src[i]);
}

// Wt[n*D + k] = W[k*D + n]  (store weight transposed, bf16)
__global__ void transpose_cvt_kernel(const float* __restrict__ W,
                                     unsigned short* __restrict__ Wt, int D) {
    int idx = blockIdx.x * blockDim.x + threadIdx.x;
    if (idx < D * D) {
        int n = idx / D, k = idx - n * D;
        Wt[idx] = f2bf(W[(size_t)k * D + n]);
    }
}

// ---------------------------------------------------------------------------
// GEMM: C[M,N] = (A[M,K](bf16) x Wt[N,K](bf16)^T + bias) * outScale
// Block = 256 threads = 8 waves; block tile 128x128; wave tile 32x64 (2x4 WMMA).
// Double-buffered LDS staging via async global->LDS copies (ASYNCcnt).
// ---------------------------------------------------------------------------
template<bool F32OUT>
__global__ __launch_bounds__(256)
void gemm_bf16_kernel(const unsigned short* __restrict__ A,
                      const unsigned short* __restrict__ Bt,
                      const float* __restrict__ bias,
                      unsigned short* __restrict__ Obf,
                      float* __restrict__ Of,
                      int M, int N, int K, float outScale)
{
    __shared__ unsigned short lA[2][128 * 32];
    __shared__ unsigned short lB[2][128 * 32];

    const int tid  = threadIdx.x;
    const int lane = tid & 31;
    const int w    = tid >> 5;      // wave 0..7
    const int wrow = w >> 1;        // 0..3  (32 rows each)
    const int wcol = w & 1;         // 0..1  (64 cols each)
    const int m0   = blockIdx.y * 128;
    const int n0   = blockIdx.x * 128;
    const int lh   = lane >> 4;     // lane half: 0|1
    const int l15  = lane & 15;

    floatx8 acc[2][4];
#pragma unroll
    for (int i = 0; i < 2; ++i)
#pragma unroll
        for (int j = 0; j < 4; ++j)
#pragma unroll
            for (int e = 0; e < 8; ++e) acc[i][j][e] = 0.f;

    // Async-stage one 128x32 tile of A and of Wt into buffer `buf`.
    auto stageAsync = [&](int buf, int kk) {
#pragma unroll
        for (int c = 0; c < 2; ++c) {
            int id  = tid + 256 * c;
            int row = id >> 2;
            int col = (id & 3) * 8;
            async_copy_b128(lds_off(&lA[buf][row * 32 + col]),
                            &A[(size_t)(m0 + row) * K + kk + col]);
            async_copy_b128(lds_off(&lB[buf][row * 32 + col]),
                            &Bt[(size_t)(n0 + row) * K + kk + col]);
        }
    };

    stageAsync(0, 0);
    wait_async0();
    __syncthreads();

    int buf = 0;
    for (int kk = 0; kk < K; kk += 32) {
        // Prefetch next K-slab into the other buffer while computing this one.
        if (kk + 32 < K) stageAsync(buf ^ 1, kk + 32);

        // A frag layout (16-bit, 16x32): lane holds row (lane&15);
        //   lanes 0-15: K {0..7,16..23}; lanes 16-31: K {8..15,24..31}
        FragBF aF[2], bF[4];
#pragma unroll
        for (int mt = 0; mt < 2; ++mt) {
            int row = wrow * 32 + mt * 16 + l15;
            int kb  = 8 * lh;
            aF[mt].q[0] = *(const uint4*)&lA[buf][row * 32 + kb];
            aF[mt].q[1] = *(const uint4*)&lA[buf][row * 32 + 16 + kb];
        }
        // B frag (32x16): lanes 0-15 col n, K {0..7,16..23}; lanes 16-31 col n, K {8..15,24..31}
#pragma unroll
        for (int nt = 0; nt < 4; ++nt) {
            int row = wcol * 64 + nt * 16 + l15;   // row of Wt == output col
            int kb  = 8 * lh;
            bF[nt].q[0] = *(const uint4*)&lB[buf][row * 32 + kb];
            bF[nt].q[1] = *(const uint4*)&lB[buf][row * 32 + 16 + kb];
        }
#pragma unroll
        for (int mt = 0; mt < 2; ++mt)
#pragma unroll
            for (int nt = 0; nt < 4; ++nt)
                acc[mt][nt] = __builtin_amdgcn_wmma_f32_16x16x32_bf16(
                    false, aF[mt].bf, false, bF[nt].bf,
                    (short)0, acc[mt][nt], false, false);

        wait_async0();      // this wave's prefetch landed
        __syncthreads();    // everyone done reading buf / writing buf^1
        buf ^= 1;
    }

    // Epilogue: C frag layout -> lanes 0-15 rows 0..7, lanes 16-31 rows 8..15
#pragma unroll
    for (int mt = 0; mt < 2; ++mt)
#pragma unroll
        for (int nt = 0; nt < 4; ++nt) {
            int col = n0 + wcol * 64 + nt * 16 + l15;
            float bv = bias[col];
#pragma unroll
            for (int r = 0; r < 8; ++r) {
                int m = m0 + wrow * 32 + mt * 16 + r + 8 * lh;
                float v = (acc[mt][nt][r] + bv) * outScale;
                if (F32OUT) Of[(size_t)m * N + col] = v;
                else        Obf[(size_t)m * N + col] = f2bf(v);
            }
        }
}

// ---------------------------------------------------------------------------
// Flash-style attention per (b,h). Block = 256 thr = 8 waves, each wave owns
// one 16-row Q tile; j loop in steps of 32 with online softmax.
// Q is pre-scaled by dk^-0.5 in its projection epilogue.
// ---------------------------------------------------------------------------
__global__ __launch_bounds__(256)
void attn_kernel(const unsigned short* __restrict__ Q,
                 const unsigned short* __restrict__ Kb,
                 const unsigned short* __restrict__ V,
                 const int* __restrict__ mask,
                 unsigned short* __restrict__ X)
{
    __shared__ unsigned short lK[32 * 64];     // K tile, row-major [j][d]
    __shared__ unsigned short lV[64 * 32];     // V tile transposed [d][j]
    __shared__ unsigned short lP[8][16][32];   // per-wave P tile (bf16)

    const int tid  = threadIdx.x;
    const int lane = tid & 31;
    const int w    = tid >> 5;
    const int lh   = lane >> 4;
    const int l15  = lane & 15;

    const int bh = blockIdx.y;
    const int b  = bh & (BATCH - 1);
    const int h  = bh >> 1;
    const int qbase = (blockIdx.x * 8 + w) * 16;
    const size_t hoff = (size_t)h * DK;

    // Q fragments: two K-chunks of 32 along dk (resident for whole kernel)
    FragBF aQ[2];
#pragma unroll
    for (int dc = 0; dc < 2; ++dc) {
        size_t ro = ((size_t)(qbase + l15) * BATCH + b) * DMODEL + hoff + dc * 32;
        int kb = 8 * lh;
        aQ[dc].q[0] = *(const uint4*)&Q[ro + kb];
        aQ[dc].q[1] = *(const uint4*)&Q[ro + 16 + kb];
    }

    floatx8 o[4];
#pragma unroll
    for (int dt = 0; dt < 4; ++dt)
#pragma unroll
        for (int e = 0; e < 8; ++e) o[dt][e] = 0.f;

    float mrow[8], lrow[8];
#pragma unroll
    for (int r = 0; r < 8; ++r) { mrow[r] = -1e30f; lrow[r] = 0.f; }

    for (int j0 = 0; j0 < S_LEN; j0 += 32) {
        __syncthreads();
        // Stage K (async, row-major) and V (register scatter -> transposed)
        {
            int jrow = tid >> 3;
            int coff = (tid & 7) * 8;
            size_t g = ((size_t)(j0 + jrow) * BATCH + b) * DMODEL + hoff + coff;
            async_copy_b128(lds_off(&lK[jrow * 64 + coff]), &Kb[g]);
            union { uint4 q; unsigned short e[8]; } vv;
            vv.q = *(const uint4*)&V[g];
#pragma unroll
            for (int e = 0; e < 8; ++e) lV[(coff + e) * 32 + jrow] = vv.e[e];
        }
        wait_async0();
        __syncthreads();

        // Scores: 16 x 32 (two 16x16 WMMA tiles, each sums two dk chunks)
        floatx8 sc[2];
#pragma unroll
        for (int jt = 0; jt < 2; ++jt) {
            floatx8 a;
#pragma unroll
            for (int e = 0; e < 8; ++e) a[e] = 0.f;
#pragma unroll
            for (int dc = 0; dc < 2; ++dc) {
                FragBF bK;
                int row = jt * 16 + l15;        // key index in tile
                int kb  = 8 * lh;
                bK.q[0] = *(const uint4*)&lK[row * 64 + dc * 32 + kb];
                bK.q[1] = *(const uint4*)&lK[row * 64 + dc * 32 + 16 + kb];
                a = __builtin_amdgcn_wmma_f32_16x16x32_bf16(
                        false, aQ[dc].bf, false, bK.bf, (short)0, a, false, false);
            }
            sc[jt] = a;
        }

        const int mk0 = mask[j0 + l15];
        const int mk1 = mask[j0 + 16 + l15];

        // Online softmax over the 32 new keys (row reductions across 16 lanes)
#pragma unroll
        for (int r = 0; r < 8; ++r) {
            float s0 = sc[0][r]; if (mk0 == 0) s0 = -1e9f;
            float s1 = sc[1][r]; if (mk1 == 0) s1 = -1e9f;
            float tm = fmaxf(s0, s1);
#pragma unroll
            for (int d = 1; d < 16; d <<= 1) tm = fmaxf(tm, __shfl_xor(tm, d, 32));
            float mn = fmaxf(mrow[r], tm);
            float alpha = __expf(mrow[r] - mn);
            float p0 = __expf(s0 - mn);
            float p1 = __expf(s1 - mn);
            float rs = p0 + p1;
#pragma unroll
            for (int d = 1; d < 16; d <<= 1) rs += __shfl_xor(rs, d, 32);
            lrow[r] = lrow[r] * alpha + rs;
            mrow[r] = mn;
#pragma unroll
            for (int dt = 0; dt < 4; ++dt) o[dt][r] *= alpha;
            int prow = r + 8 * lh;              // C-frag row -> P row
            lP[w][prow][l15]      = f2bf(p0);
            lP[w][prow][16 + l15] = f2bf(p1);
        }

        // P(16x32) @ V(32x64): A-frag from LDS P, B-frag from transposed V
        FragBF pF;
        {
            int kb = 8 * lh;
            pF.q[0] = *(const uint4*)&lP[w][l15][kb];
            pF.q[1] = *(const uint4*)&lP[w][l15][16 + kb];
        }
#pragma unroll
        for (int dt = 0; dt < 4; ++dt) {
            FragBF bV;
            int d  = dt * 16 + l15;             // output feature column
            int kb = 8 * lh;
            bV.q[0] = *(const uint4*)&lV[d * 32 + kb];
            bV.q[1] = *(const uint4*)&lV[d * 32 + 16 + kb];
            o[dt] = __builtin_amdgcn_wmma_f32_16x16x32_bf16(
                        false, pF.bf, false, bV.bf, (short)0, o[dt], false, false);
        }
    }

    // Normalize and store x (bf16) for the output projection
#pragma unroll
    for (int dt = 0; dt < 4; ++dt)
#pragma unroll
        for (int r = 0; r < 8; ++r) {
            int row = qbase + r + 8 * lh;
            float linv = lrow[r] > 0.f ? 1.f / lrow[r] : 0.f;
            X[((size_t)row * BATCH + b) * DMODEL + hoff + dt * 16 + l15] =
                f2bf(o[dt][r] * linv);
        }
}

// ---------------------------------------------------------------------------
// Host-side orchestration
// ---------------------------------------------------------------------------
extern "C" void kernel_launch(void* const* d_in, const int* in_sizes, int n_in,
                              void* d_out, int out_size, void* d_ws, size_t ws_size,
                              hipStream_t stream) {
    (void)in_sizes; (void)n_in; (void)out_size; (void)ws_size;

    const float* q_in = (const float*)d_in[0];
    const float* k_in = (const float*)d_in[1];
    const float* v_in = (const float*)d_in[2];
    const int*   mask = (const int*)  d_in[3];
    const float* Wq   = (const float*)d_in[4];
    const float* bq   = (const float*)d_in[5];
    const float* Wk   = (const float*)d_in[6];
    const float* bk   = (const float*)d_in[7];
    const float* Wv   = (const float*)d_in[8];
    const float* bv   = (const float*)d_in[9];
    const float* Wo   = (const float*)d_in[10];
    const float* bo   = (const float*)d_in[11];
    float* out = (float*)d_out;

    const int D  = DMODEL;
    const int MB = S_LEN * BATCH;          // 4096 tokens
    const size_t inE = (size_t)MB * D;     // 4 Mi elements
    const size_t wE  = (size_t)D * D;      // 1 Mi elements

    char* p = (char*)d_ws;
    unsigned short* qin = (unsigned short*)p; p += inE * 2;
    unsigned short* kin = (unsigned short*)p; p += inE * 2;
    unsigned short* vin = (unsigned short*)p; p += inE * 2;
    unsigned short* wqT = (unsigned short*)p; p += wE * 2;
    unsigned short* wkT = (unsigned short*)p; p += wE * 2;
    unsigned short* wvT = (unsigned short*)p; p += wE * 2;
    unsigned short* woT = (unsigned short*)p; p += wE * 2;
    unsigned short* qp  = (unsigned short*)p; p += inE * 2;
    unsigned short* kp  = (unsigned short*)p; p += inE * 2;
    unsigned short* vp  = (unsigned short*)p; p += inE * 2;
    unsigned short* xbf = (unsigned short*)p; p += inE * 2;

    const int thr = 256;
    const int gIn = (int)((inE + thr - 1) / thr);
    const int gW  = (int)((wE + thr - 1) / thr);

    cvt_bf16_kernel<<<gIn, thr, 0, stream>>>(q_in, qin, (int)inE);
    cvt_bf16_kernel<<<gIn, thr, 0, stream>>>(k_in, kin, (int)inE);
    cvt_bf16_kernel<<<gIn, thr, 0, stream>>>(v_in, vin, (int)inE);
    transpose_cvt_kernel<<<gW, thr, 0, stream>>>(Wq, wqT, D);
    transpose_cvt_kernel<<<gW, thr, 0, stream>>>(Wk, wkT, D);
    transpose_cvt_kernel<<<gW, thr, 0, stream>>>(Wv, wvT, D);
    transpose_cvt_kernel<<<gW, thr, 0, stream>>>(Wo, woT, D);

    const float qscale = 0.125f;           // dk^-0.5 baked into Q projection
    dim3 gg(D / 128, MB / 128);            // (8, 32)
    gemm_bf16_kernel<false><<<gg, thr, 0, stream>>>(qin, wqT, bq, qp, nullptr, MB, D, D, qscale);
    gemm_bf16_kernel<false><<<gg, thr, 0, stream>>>(kin, wkT, bk, kp, nullptr, MB, D, D, 1.0f);
    gemm_bf16_kernel<false><<<gg, thr, 0, stream>>>(vin, wvT, bv, vp, nullptr, MB, D, D, 1.0f);

    dim3 ga(S_LEN / (16 * 8), BATCH * NHEAD);  // (16, 32)
    attn_kernel<<<ga, thr, 0, stream>>>(qp, kp, vp, mask, xbf);

    gemm_bf16_kernel<true><<<gg, thr, 0, stream>>>(xbf, woT, bo, nullptr, out, MB, D, D, 1.0f);
}
